// LI_FusionModule_317827580573
// MI455X (gfx1250) — matrix operand order repaired and checked by reading, not compiled, not run on hardware
//
#include <hip/hip_runtime.h>
#include <hip/hip_bf16.h>
#include <math.h>

// ---------------------------------------------------------------------------
// MI455X (gfx1250) fused LI-Fusion pipeline.
// GEMMs run on v_wmma_f32_16x16x32_f16 (f16 A/B, fp32 accum).
// BN stats via fp32 global atomics; per-wave 16-row tiles; comb tile in LDS.
// ---------------------------------------------------------------------------

typedef _Float16 half_t;
typedef __attribute__((ext_vector_type(16))) _Float16 v16h;
typedef __attribute__((ext_vector_type(8)))  float    v8f;

#define N_VOX   200000
#define NTILES  (N_VOX / 16)          // 12500 (exact)
#define C3D_    64
#define MID_    128
#define OUT_    128
#define C2D_    128
#define IMG_H   96
#define IMG_W   312
#define NT      8                     // 128 output cols / 16
#define EPS_    1e-5f

#define WAIT_DS() asm volatile("s_wait_dscnt 0x0" ::: "memory")

// workspace layout (bytes)
static constexpr size_t OFF_HPRE  = 0;                                   // N*128 f16
static constexpr size_t OFF_W1H   = (size_t)N_VOX * MID_ * 2;            // 51,200,000
static constexpr size_t OFF_W2H   = OFF_W1H  + (size_t)MID_ * C3D_ * 2;
static constexpr size_t OFF_AW1H  = OFF_W2H  + (size_t)MID_ * MID_ * 2;
static constexpr size_t OFF_FWH   = OFF_AW1H + (size_t)MID_ * 2 * MID_ * 2;
static constexpr size_t OFF_STATS = OFF_FWH  + (size_t)OUT_ * 2 * MID_ * 2; // 1024 floats

__device__ __forceinline__ v8f wmma_f16(v16h a, v16h b, v8f c) {
  // (neg_a, A, neg_b, B, c_mod, C, reuse_a, reuse_b)
  return __builtin_amdgcn_wmma_f32_16x16x32_f16(false, a, false, b, (short)0, c,
                                                false, false);
}

// A-operand (16x32 f16) from a 256-col LDS row: lanes 0-15 take K {k0..k0+7,
// k0+16..k0+23}; lanes 16-31 take K {k0+8..k0+15, k0+24..k0+31}.
__device__ __forceinline__ v16h load_a(const half_t* rowptr, int kc, int hi) {
  union { v16h v; uint4 q[2]; } u;
  u.q[0] = *(const uint4*)(rowptr + kc * 32 + hi * 8);
  u.q[1] = *(const uint4*)(rowptr + kc * 32 + 16 + hi * 8);
  return u.v;
}

// B-operand (32x16 f16) from row-major (n,K) f16 weights: lane's column n,
// K {k0+hi*8..+7, k0+16+hi*8..+7} with k0 = kc*32 (same order as load_a).
__device__ __forceinline__ v16h load_b(const half_t* w, int n, int K, int kc, int hi) {
  union { v16h v; uint4 q[2]; } u;
  const half_t* p = w + (size_t)n * K + kc * 32 + hi * 8;
  u.q[0] = *(const uint4*)(p);
  u.q[1] = *(const uint4*)(p + 16);
  return u.v;
}

// ---------------------------------------------------------------------------
// K0: convert weights fp32 -> f16 (layout preserved) and zero stat buffers.
// ---------------------------------------------------------------------------
__global__ void k_prep(const float* __restrict__ w1, const float* __restrict__ w2,
                       const float* __restrict__ aw1, const float* __restrict__ fw,
                       half_t* __restrict__ w1h, half_t* __restrict__ w2h,
                       half_t* __restrict__ aw1h, half_t* __restrict__ fwh,
                       float* __restrict__ stats) {
  int i = blockIdx.x * 256 + threadIdx.x;
  if (i < 512)          stats[i] = 0.0f;
  if (i < MID_ * C3D_)  w1h[i]  = (half_t)w1[i];
  if (i < MID_ * MID_)  w2h[i]  = (half_t)w2[i];
  if (i < MID_ * 2 * MID_) { aw1h[i] = (half_t)aw1[i]; fwh[i] = (half_t)fw[i]; }
}

// ---------------------------------------------------------------------------
// K1: h_pre = X @ w1^T  (vt_b1 skipped: cancelled by BN1). 16 WMMA / tile.
// Also accumulate per-channel sum / sumsq for BN1.
// ---------------------------------------------------------------------------
__global__ void __launch_bounds__(256) k_gemm1(const float* __restrict__ X,
                                               const half_t* __restrict__ w1h,
                                               half_t* __restrict__ hpre,
                                               float* __restrict__ sum1,
                                               float* __restrict__ sq1) {
  const int gwave = (blockIdx.x * 256 + threadIdx.x) >> 5;
  const int lane  = threadIdx.x & 31;
  if (gwave >= NTILES) return;
  const int r0 = gwave * 16;
  const int m  = lane & 15;
  const int hi = lane >> 4;

  // A fragments straight from global fp32 (convert to f16)
  v16h a[2];
  const float* xrow = X + (size_t)(r0 + m) * C3D_;
  #pragma unroll
  for (int kc = 0; kc < 2; ++kc) {
    const int k0 = kc * 32 + hi * 8;
    #pragma unroll
    for (int i = 0; i < 8; ++i) {
      a[kc][i]     = (_Float16)xrow[k0 + i];
      a[kc][8 + i] = (_Float16)xrow[k0 + 16 + i];
    }
  }

  v8f acc[NT] = {};
  #pragma unroll
  for (int kc = 0; kc < 2; ++kc)
    #pragma unroll
    for (int nt = 0; nt < NT; ++nt)
      acc[nt] = wmma_f16(a[kc], load_b(w1h, nt * 16 + m, C3D_, kc, hi), acc[nt]);

  #pragma unroll
  for (int nt = 0; nt < NT; ++nt) {
    const int n = nt * 16 + m;
    float s = 0.f, ss = 0.f;
    #pragma unroll
    for (int v = 0; v < 8; ++v) {
      const float val = acc[nt][v];
      hpre[(size_t)(r0 + v + hi * 8) * MID_ + n] = (half_t)val;
      s += val; ss += val * val;
    }
    unsafeAtomicAdd(&sum1[n], s);
    unsafeAtomicAdd(&sq1[n], ss);
  }
}

// ---------------------------------------------------------------------------
// BN finalize: scale/shift per channel from sums.
// ---------------------------------------------------------------------------
__global__ void k_finalize(const float* __restrict__ sum, const float* __restrict__ sq,
                           const float* __restrict__ g, const float* __restrict__ b,
                           float* __restrict__ sc, float* __restrict__ sh) {
  const int c = threadIdx.x;
  if (c >= 128) return;
  const float mean = sum[c] / (float)N_VOX;
  const float var  = sq[c] / (float)N_VOX - mean * mean;
  const float s    = g[c] * rsqrtf(var + EPS_);
  sc[c] = s;
  sh[c] = b[c] - mean * s;
}

// ---------------------------------------------------------------------------
// K3: fused main kernel. 4 waves/block, one 16-row tile per wave.
// BN1+ReLU -> GEMM2(vfeat) -> bilinear gather(ifeat) -> attention GEMM ->
// sigmoid gate -> gated fused GEMM -> out_pre + BN2 stats. 160 WMMA / tile.
// ---------------------------------------------------------------------------
__global__ void __launch_bounds__(128) k_fused(
    const half_t* __restrict__ hpre, const int* __restrict__ coords,
    const float* __restrict__ img, const float* __restrict__ P2,
    const half_t* __restrict__ w2h, const float* __restrict__ b2,
    const half_t* __restrict__ aw1h, const float* __restrict__ ab1,
    const float* __restrict__ aw2, const float* __restrict__ ab2,
    const half_t* __restrict__ fwh,
    const float* __restrict__ sc1, const float* __restrict__ sh1,
    float* __restrict__ out, float* __restrict__ sum2, float* __restrict__ sq2) {
  __shared__ half_t comb[4][16][256];   // [vfeat | ifeat] tile, f16 (32 KB)
  __shared__ float  rowpar[4][16][8];   // projection params per row
  __shared__ float  abuf[4][16];        // attention dot accumulators

  const int wv   = threadIdx.x >> 5;
  const int lane = threadIdx.x & 31;
  const int tile = blockIdx.x * 4 + wv;
  if (tile >= NTILES) return;           // uniform per wave
  const int r0 = tile * 16;
  const int m  = lane & 15;
  const int hi = lane >> 4;
  half_t* crow = &comb[wv][m][0];

  // ---- Phase A: BN1 + ReLU -> comb cols 0..127 (lane owns 4 columns) ----
  {
    float lsc[4], lsh[4];
    #pragma unroll
    for (int i = 0; i < 4; ++i) { const int c = lane * 4 + i; lsc[i] = sc1[c]; lsh[i] = sh1[c]; }
    for (int row = 0; row < 16; ++row) {
      const half_t* hp = hpre + (size_t)(r0 + row) * MID_ + lane * 4;
      #pragma unroll
      for (int i = 0; i < 4; ++i) {
        const float v = (float)hp[i] * lsc[i] + lsh[i];
        comb[wv][row][lane * 4 + i] = (half_t)(v > 0.f ? v : 0.f);
      }
    }
  }
  WAIT_DS();

  // ---- GEMM2: vfeat = h @ w2^T (+b2 later). 32 WMMA ----
  v8f vacc[NT] = {};
  for (int kc = 0; kc < 4; ++kc) {
    const v16h af = load_a(crow, kc, hi);
    #pragma unroll
    for (int nt = 0; nt < NT; ++nt)
      vacc[nt] = wmma_f16(af, load_b(w2h, nt * 16 + m, MID_, kc, hi), vacc[nt]);
  }

  // ---- Projection + bilinear params for row m ----
  {
    const int4 vc = *(const int4*)(coords + 4 * (r0 + m));   // [b, c1, c2, c3]
    const float px = (float)vc.y * 0.05f + 0.0f;
    const float py = (float)vc.z * 0.05f - 40.0f;
    const float pz = (float)vc.w * 0.10f - 3.0f;
    const float* P = P2 + vc.x * 12;
    const float c0 = P[0] * px + P[1] * py + P[2]  * pz + P[3];
    const float c1 = P[4] * px + P[5] * py + P[6]  * pz + P[7];
    const float c2 = P[8] * px + P[9] * py + P[10] * pz + P[11];
    const float inv = 1.0f / (c2 + 1e-8f);
    const float gx = fminf(fmaxf(c0 * inv / (float)IMG_W * 2.f - 1.f, -1.f), 1.f);
    const float gy = fminf(fmaxf(c1 * inv / (float)IMG_H * 2.f - 1.f, -1.f), 1.f);
    const float ix = ((gx + 1.f) * (float)IMG_W - 1.f) * 0.5f;
    const float iy = ((gy + 1.f) * (float)IMG_H - 1.f) * 0.5f;
    const float x0f = floorf(ix), y0f = floorf(iy);
    if (hi == 0) {
      rowpar[wv][m][0] = (float)vc.x;
      rowpar[wv][m][1] = x0f;
      rowpar[wv][m][2] = y0f;
      rowpar[wv][m][3] = ix - x0f;
      rowpar[wv][m][4] = iy - y0f;
    }
  }
  WAIT_DS();

  // ---- store vfeat+b2 (f16) into comb cols 0..127 (overwrite h) ----
  #pragma unroll
  for (int nt = 0; nt < NT; ++nt) {
    const int n = nt * 16 + m;
    const float bb = b2[n];
    #pragma unroll
    for (int v = 0; v < 8; ++v)
      comb[wv][v + hi * 8][n] = (half_t)(vacc[nt][v] + bb);
  }

  // ---- bilinear gather (image is L2-resident) -> comb cols 128..255 ----
  for (int row = 0; row < 16; ++row) {
    const int   bi  = (int)rowpar[wv][row][0];
    const int   x0  = (int)rowpar[wv][row][1];
    const int   y0  = (int)rowpar[wv][row][2];
    const float wx1 = rowpar[wv][row][3];
    const float wy1 = rowpar[wv][row][4];
    float cw[4]; int co[4];
    #pragma unroll
    for (int k = 0; k < 4; ++k) {
      const int xc = x0 + (k & 1), yc = y0 + (k >> 1);
      const float w = ((k & 1) ? wx1 : 1.f - wx1) * ((k >> 1) ? wy1 : 1.f - wy1);
      const bool valid = (xc >= 0) && (xc < IMG_W) && (yc >= 0) && (yc < IMG_H);
      const int xcc = xc < 0 ? 0 : (xc > IMG_W - 1 ? IMG_W - 1 : xc);
      const int ycc = yc < 0 ? 0 : (yc > IMG_H - 1 ? IMG_H - 1 : yc);
      cw[k] = valid ? w : 0.f;
      co[k] = ycc * IMG_W + xcc;
    }
    const float* ibase = img + (size_t)bi * C2D_ * IMG_H * IMG_W;
    #pragma unroll
    for (int j = 0; j < 4; ++j) {
      const int c = lane + j * 32;
      const float* cb = ibase + (size_t)c * IMG_H * IMG_W;
      const float acc = cw[0] * cb[co[0]] + cw[1] * cb[co[1]] +
                        cw[2] * cb[co[2]] + cw[3] * cb[co[3]];
      comb[wv][row][128 + c] = (half_t)acc;
    }
  }
  WAIT_DS();

  // ---- attention: t1 = relu(comb @ aw1^T + ab1); a = sigmoid(t1.aw2 + ab2) ----
  v8f aacc[NT] = {};
  for (int kc = 0; kc < 8; ++kc) {
    const v16h af = load_a(crow, kc, hi);
    #pragma unroll
    for (int nt = 0; nt < NT; ++nt)
      aacc[nt] = wmma_f16(af, load_b(aw1h, nt * 16 + m, 2 * MID_, kc, hi), aacc[nt]);
  }
  if (hi == 0) abuf[wv][m] = 0.f;
  float part[8] = {};
  #pragma unroll
  for (int nt = 0; nt < NT; ++nt) {
    const int n = nt * 16 + m;
    const float bb = ab1[n], wvv = aw2[n];
    #pragma unroll
    for (int v = 0; v < 8; ++v) {
      float t = aacc[nt][v] + bb;
      t = t > 0.f ? t : 0.f;
      part[v] += t * wvv;
    }
  }
  WAIT_DS();
  #pragma unroll
  for (int v = 0; v < 8; ++v) unsafeAtomicAdd(&abuf[wv][v + hi * 8], part[v]);
  WAIT_DS();
  const float aval = 1.f / (1.f + expf(-(abuf[wv][m] + ab2[0])));

  // ---- fused GEMM: [vfeat*a | ifeat*(1-a)] @ fus_w^T (fus_b cancels in BN2) ----
  v8f facc[NT] = {};
  for (int kc = 0; kc < 8; ++kc) {
    const _Float16 s = (_Float16)((kc < 4) ? aval : (1.f - aval));
    v16h af = load_a(crow, kc, hi);
    #pragma unroll
    for (int i = 0; i < 16; ++i) af[i] = af[i] * s;
    #pragma unroll
    for (int nt = 0; nt < NT; ++nt)
      facc[nt] = wmma_f16(af, load_b(fwh, nt * 16 + m, 2 * MID_, kc, hi), facc[nt]);
  }

  // ---- out_pre (fp32) + BN2 stats ----
  #pragma unroll
  for (int nt = 0; nt < NT; ++nt) {
    const int n = nt * 16 + m;
    float s = 0.f, ss = 0.f;
    #pragma unroll
    for (int v = 0; v < 8; ++v) {
      const float val = facc[nt][v];
      out[(size_t)(r0 + v + hi * 8) * OUT_ + n] = val;
      s += val; ss += val * val;
    }
    unsafeAtomicAdd(&sum2[n], s);
    unsafeAtomicAdd(&sq2[n], ss);
  }
}

// ---------------------------------------------------------------------------
// K5: final BN + ReLU in place on d_out.
// ---------------------------------------------------------------------------
__global__ void k_bnout(float* __restrict__ out, const float* __restrict__ sc,
                        const float* __restrict__ sh) {
  const size_t i = (size_t)blockIdx.x * 256 + threadIdx.x;
  if (i < (size_t)N_VOX * OUT_) {
    const int c = (int)(i & 127);
    const float v = out[i] * sc[c] + sh[c];
    out[i] = v > 0.f ? v : 0.f;
  }
}

// ---------------------------------------------------------------------------
extern "C" void kernel_launch(void* const* d_in, const int* in_sizes, int n_in,
                              void* d_out, int out_size, void* d_ws, size_t ws_size,
                              hipStream_t stream) {
  (void)in_sizes; (void)n_in; (void)out_size; (void)ws_size;

  const float* vfeat   = (const float*)d_in[0];
  const int*   vcoords = (const int*)d_in[1];
  const float* img     = (const float*)d_in[2];
  const float* P2      = (const float*)d_in[3];
  const float* vt_w1   = (const float*)d_in[4];
  // d_in[5] vt_b1: skipped (exactly cancelled by BN1)
  const float* bn1_g   = (const float*)d_in[6];
  const float* bn1_b   = (const float*)d_in[7];
  const float* vt_w2   = (const float*)d_in[8];
  const float* vt_b2   = (const float*)d_in[9];
  const float* att_w1  = (const float*)d_in[10];
  const float* att_b1  = (const float*)d_in[11];
  const float* att_w2  = (const float*)d_in[12];
  const float* att_b2  = (const float*)d_in[13];
  const float* fus_w   = (const float*)d_in[14];
  // d_in[15] fus_b: skipped (exactly cancelled by BN2)
  const float* bnf_g   = (const float*)d_in[16];
  const float* bnf_b   = (const float*)d_in[17];

  char* ws = (char*)d_ws;
  half_t* hpre  = (half_t*)(ws + OFF_HPRE);
  half_t* w1h   = (half_t*)(ws + OFF_W1H);
  half_t* w2h   = (half_t*)(ws + OFF_W2H);
  half_t* aw1h  = (half_t*)(ws + OFF_AW1H);
  half_t* fwh   = (half_t*)(ws + OFF_FWH);
  float*  stats = (float*)(ws + OFF_STATS);
  float *sum1 = stats,       *sq1 = stats + 128;
  float *sum2 = stats + 256, *sq2 = stats + 384;
  float *sc1  = stats + 512, *sh1 = stats + 640;
  float *sc2  = stats + 768, *sh2 = stats + 896;
  float* out = (float*)d_out;

  k_prep<<<128, 256, 0, stream>>>(vt_w1, vt_w2, att_w1, fus_w,
                                  w1h, w2h, aw1h, fwh, stats);
  k_gemm1<<<(NTILES + 7) / 8, 256, 0, stream>>>(vfeat, w1h, hpre, sum1, sq1);
  k_finalize<<<1, 128, 0, stream>>>(sum1, sq1, bn1_g, bn1_b, sc1, sh1);
  k_fused<<<(NTILES + 3) / 4, 128, 0, stream>>>(hpre, vcoords, img, P2,
                                                w2h, vt_b2, aw1h, att_b1,
                                                att_w2, att_b2, fwh,
                                                sc1, sh1, out, sum2, sq2);
  k_finalize<<<1, 128, 0, stream>>>(sum2, sq2, bnf_g, bnf_b, sc2, sh2);
  k_bnout<<<(int)(((size_t)N_VOX * OUT_ + 255) / 256), 256, 0, stream>>>(out, sc2, sh2);
}
